// BiLSTM_classifier_13623636263489
// MI455X (gfx1250) — compile-verified, matrix operand order checked by reading, not run on hardware
//
#include <hip/hip_runtime.h>
#include <hip/hip_bf16.h>

#define TLEN 8192
#define ED   512
#define HD   512
#define H4   2048
#define NC   2

typedef __attribute__((ext_vector_type(16))) __bf16 v16bf;
typedef __attribute__((ext_vector_type(8)))  __bf16 v8bf;
typedef __attribute__((ext_vector_type(8)))  float  v8f;
typedef int v4i_gcc __attribute__((vector_size(16)));   // matches builtin "V4i"

#define AS1 __attribute__((address_space(1)))
#define AS3 __attribute__((address_space(3)))

// ---- CDNA5 async global->LDS copy (ASYNCcnt-tracked), with safe fallback ----
#if defined(__has_builtin)
#if __has_builtin(__builtin_amdgcn_global_load_async_to_lds_b128)
#define HAVE_ASYNC_LDS 1
#endif
#endif

#ifdef HAVE_ASYNC_LDS
// Signature (from clang diagnostic): (global v4i*, shared v4i*, imm offset, imm cpol)
#define ASYNC_B128(gp, lp)                                                     \
    __builtin_amdgcn_global_load_async_to_lds_b128(                            \
        (AS1 v4i_gcc*)(size_t)(gp), (AS3 v4i_gcc*)(unsigned)(size_t)(lp), 0, 0)
#if __has_builtin(__builtin_amdgcn_s_wait_asynccnt)
#define WAIT_ASYNC(n) __builtin_amdgcn_s_wait_asynccnt(n)
#else
#define WAIT_ASYNC(n) asm volatile("s_wait_asynccnt %0" ::"i"(n) : "memory")
#endif
#endif

// ---------- bf16 helpers (bit-exact RNE, no reliance on hw cvt) ----------
__device__ __forceinline__ unsigned short f2bf_u16(float f) {
    unsigned u = __float_as_uint(f);
    return (unsigned short)((u + 0x7FFFu + ((u >> 16) & 1u)) >> 16);
}
__device__ __forceinline__ __bf16 f2bf(float f) {
    unsigned short s = f2bf_u16(f);
    __bf16 r; __builtin_memcpy(&r, &s, 2); return r;
}

// Load one lane's 16-element bf16 WMMA fragment slice for a 16x32 (MxK) tile.
// Layout (ISA 7.12.2, 16-bit A 16x32): lanes 0-15 hold K {0..7,16..23},
// lanes 16-31 hold K {8..15,24..31}. Works for global or LDS source pointers.
__device__ __forceinline__ v16bf load_frag_row(const __bf16* rowk, int lane) {
    const __bf16* p = rowk + (((lane >> 4) & 1) << 3);   // +0 or +8
    v8bf lo = *(const v8bf*)p;
    v8bf hi = *(const v8bf*)(p + 16);
    return __builtin_shufflevector(lo, hi, 0,1,2,3,4,5,6,7,8,9,10,11,12,13,14,15);
}

// ------------------------- prep kernels -------------------------
__global__ void f32_to_bf16_kernel(const float* __restrict__ src,
                                   __bf16* __restrict__ dst, int n) {
    int i = blockIdx.x * blockDim.x + threadIdx.x;
    int stride = gridDim.x * blockDim.x;
    for (; i < n; i += stride) dst[i] = f2bf(src[i]);
}

__global__ void bias_sum_kernel(const float* __restrict__ a,
                                const float* __restrict__ b,
                                float* __restrict__ dst, int n) {
    int i = blockIdx.x * blockDim.x + threadIdx.x;
    if (i < n) dst[i] = a[i] + b[i];
}

__global__ void zero_bf16_kernel(__bf16* __restrict__ p, int n) {
    int i = blockIdx.x * blockDim.x + threadIdx.x;
    if (i < n) p[i] = f2bf(0.0f);
}

// Embedding gather: one block per token row, f32 -> bf16 packed stores.
__global__ void __launch_bounds__(128)
gather_embed_kernel(const int* __restrict__ idx, const float* __restrict__ emb,
                    __bf16* __restrict__ X) {
    int row = blockIdx.x;
    int c = threadIdx.x * 4;
    int r = idx[row];
    float4 v = *(const float4*)(emb + (size_t)r * ED + c);
    unsigned lo = (unsigned)f2bf_u16(v.x) | ((unsigned)f2bf_u16(v.y) << 16);
    unsigned hi = (unsigned)f2bf_u16(v.z) | ((unsigned)f2bf_u16(v.w) << 16);
    uint2 u; u.x = lo; u.y = hi;
    *(uint2*)(X + (size_t)row * ED + c) = u;
}

// ------------------------- x-projection GEMM -------------------------
// XP[M=8192, N=2048] = X[8192x512](bf16) @ W^T (W = Wih [2048x512] bf16) + bsum
// Block = 256 threads = 8 waves (4x2); block tile 128(M) x 64(N).
// Async path: double-buffered LDS staging via global_load_async_to_lds_b128
// (ASYNCcnt), WMMA fragments fed from LDS (ds_load_b128).
__global__ void __launch_bounds__(256)
xproj_gemm_kernel(const __bf16* __restrict__ X, const __bf16* __restrict__ W,
                  const float* __restrict__ bsum, float* __restrict__ XP,
                  int rev) {
    int tid  = threadIdx.x;
    int lane = tid & 31;
    int w    = tid >> 5;              // 0..7
    int wm   = w & 3;                 // 4 waves along M
    int wn   = w >> 2;                // 2 waves along N
    int m0   = blockIdx.y * 128;
    int n0   = blockIdx.x * 64;
    int lr   = lane & 15;

    v8f acc[2][2] = {};

#ifdef HAVE_ASYNC_LDS
    __shared__ __bf16 Abuf[2][128 * 32];   // 8 KB per buffer
    __shared__ __bf16 Bbuf[2][64 * 32];    // 4 KB per buffer

    // Stage one 32-wide K chunk into LDS buffer `buf` (3 async b128 per wave).
    auto stage = [&](int kc, int buf) {
        int kb = kc * 32;
#pragma unroll
        for (int s = 0; s < 2; ++s) {
            int chunk = tid + s * 256;        // 0..511 : A tile 16B chunks
            int row = chunk >> 2, q = chunk & 3;
            int m = m0 + row;
            int arow = rev ? (TLEN - 1 - m) : m;
            ASYNC_B128(X + (size_t)arow * ED + kb + q * 8, &Abuf[buf][chunk * 8]);
        }
        {
            int row = tid >> 2, q = tid & 3;  // 0..255 : B tile 16B chunks
            int n = n0 + row;
            ASYNC_B128(W + (size_t)n * ED + kb + q * 8, &Bbuf[buf][tid * 8]);
        }
    };

    stage(0, 0);
    for (int kc = 0; kc < 16; ++kc) {
        int cur = kc & 1;
        if (kc < 15) stage(kc + 1, cur ^ 1);
        if (kc < 15) { WAIT_ASYNC(3); } else { WAIT_ASYNC(0); }
        __syncthreads();
        v16bf afr[2], bfr[2];
#pragma unroll
        for (int i = 0; i < 2; ++i) {
            afr[i] = load_frag_row(&Abuf[cur][(wm * 32 + i * 16 + lr) * 32], lane);
            bfr[i] = load_frag_row(&Bbuf[cur][(wn * 32 + i * 16 + lr) * 32], lane);
        }
#pragma unroll
        for (int i = 0; i < 2; ++i)
#pragma unroll
            for (int j = 0; j < 2; ++j)
                acc[i][j] = __builtin_amdgcn_wmma_f32_16x16x32_bf16(
                    false, afr[i], false, bfr[j], (short)0, acc[i][j],
                    false, false);
        __syncthreads();   // protect cur buffer before it is re-staged
    }
#else
    // Fallback: direct global fragment loads.
    for (int kb = 0; kb < ED; kb += 32) {
        v16bf afr[2], bfr[2];
#pragma unroll
        for (int i = 0; i < 2; ++i) {
            int m = m0 + wm * 32 + i * 16 + lr;
            int arow = rev ? (TLEN - 1 - m) : m;
            const __bf16* ap = X + (size_t)arow * ED + kb;
            __builtin_prefetch(ap + 64, 0, 0);
            afr[i] = load_frag_row(ap, lane);
            int n = n0 + wn * 32 + i * 16 + lr;
            bfr[i] = load_frag_row(W + (size_t)n * ED + kb, lane);
        }
#pragma unroll
        for (int i = 0; i < 2; ++i)
#pragma unroll
            for (int j = 0; j < 2; ++j)
                acc[i][j] = __builtin_amdgcn_wmma_f32_16x16x32_bf16(
                    false, afr[i], false, bfr[j], (short)0, acc[i][j],
                    false, false);
    }
#endif

    // Epilogue per C/D layout: lane&15 = N col, (lane>>4)*8 + r = M row.
    int mbase = (lane >> 4) * 8;
#pragma unroll
    for (int i = 0; i < 2; ++i)
#pragma unroll
        for (int j = 0; j < 2; ++j) {
            int n = n0 + wn * 32 + j * 16 + lr;
            float bb = bsum[n];
#pragma unroll
            for (int r = 0; r < 8; ++r) {
                int m = m0 + wm * 32 + i * 16 + mbase + r;
                XP[(size_t)m * H4 + n] = acc[i][j][r] + bb;
            }
        }
}

// ------------------------- bidirectional scan -------------------------
// 16 workgroups = one CDNA5 cluster: blocks 0..7 forward, 8..15 backward.
// Block owns 64 h-rows (4 gate quadrants x 64 rows = 256 gate rows).
// Per step: h_{t-1} is async-staged into LDS once, matvec runs as WMMA with
// B = h broadcast into 16 cols (col 0 used), elementwise LSTM cell, publish
// h (bf16) to the double-buffered global exchange, then cluster split-barrier
// (s_barrier_signal -3 / s_barrier_wait -3).
__global__ void __launch_bounds__(256)
bilstm_scan_kernel(const __bf16* __restrict__ Whh_f, const __bf16* __restrict__ Whh_b,
                   const float* __restrict__ XPf, const float* __restrict__ XPb,
                   float* __restrict__ h_fwd, float* __restrict__ h_bwd,
                   __bf16* __restrict__ hbuf /* [dir][parity][512] */) {
    __shared__ float  gates[256];
    __shared__ float  c_st[64];
    __shared__ __bf16 hprev_lds[512];

    int tid  = threadIdx.x;
    int lane = tid & 31;
    int w    = tid >> 5;                 // 0..7
    int dir  = blockIdx.x >> 3;          // 0 fwd, 1 bwd
    int slab = blockIdx.x & 7;           // which 64 h-rows
    int lr   = lane & 15;

    const __bf16* Whh = dir ? Whh_b : Whh_f;
    const float*  XP  = dir ? XPb   : XPf;
    float*        hs  = dir ? h_bwd : h_fwd;
    __bf16*       hb  = hbuf + dir * 1024;   // two 512-wide parity buffers

    if (tid < 64) c_st[tid] = 0.0f;
    __syncthreads();

    int tt0 = w * 2;   // this wave's two 16-row gate tiles (of 16 total)
    for (int t = 0; t < TLEN; ++t) {
        const __bf16* hprev_g = hb + (t & 1) * 512;
        // Stage h_{t-1} (1 KB) into LDS: 64 x b128 chunks.
#ifdef HAVE_ASYNC_LDS
        if (tid < 64) ASYNC_B128(hprev_g + tid * 8, &hprev_lds[tid * 8]);
        WAIT_ASYNC(0);
#else
        if (tid < 64) {
            uint2 u = *(const uint2*)(hprev_g + tid * 4);
            *(uint2*)(&hprev_lds[tid * 4]) = u;
        }
#endif
        __syncthreads();

        v8f acc[2] = {};
        for (int kb = 0; kb < HD; kb += 32) {
            v16bf bfr = load_frag_row(hprev_lds + kb, lane);   // broadcast h
#pragma unroll
            for (int i = 0; i < 2; ++i) {
                int tt = tt0 + i;
                int q  = tt >> 2;        // gate quadrant (i,f,g,o)
                int rt = tt & 3;         // 16-row tile within quadrant slab
                int grow = q * 512 + slab * 64 + rt * 16 + lr;
                v16bf afr = load_frag_row(Whh + (size_t)grow * HD + kb, lane);
                acc[i] = __builtin_amdgcn_wmma_f32_16x16x32_bf16(
                    false, afr, false, bfr, (short)0, acc[i], false, false);
            }
        }
        // Column 0 of D lives in lanes 0 (rows 0..7) and 16 (rows 8..15).
        if (lr == 0) {
            int mb = (lane >> 4) * 8;
#pragma unroll
            for (int i = 0; i < 2; ++i) {
                int tt = tt0 + i;
                int q = tt >> 2, rt = tt & 3;
                int base = q * 64 + rt * 16 + mb;
#pragma unroll
                for (int r = 0; r < 8; ++r) gates[base + r] = acc[i][r];
            }
        }
        __syncthreads();

        if (tid < 64) {
            int row = slab * 64 + tid;
            const float* xp = XP + (size_t)t * H4;
            float gi = gates[tid]        + xp[row];
            float gf = gates[64 + tid]   + xp[512 + row];
            float gg = gates[128 + tid]  + xp[1024 + row];
            float go = gates[192 + tid]  + xp[1536 + row];
            float iv = 1.0f / (1.0f + __expf(-gi));
            float fv = 1.0f / (1.0f + __expf(-gf));
            float gv = tanhf(gg);
            float ov = 1.0f / (1.0f + __expf(-go));
            float c  = fv * c_st[tid] + iv * gv;
            c_st[tid] = c;
            float h = ov * tanhf(c);
            int tout = dir ? (TLEN - 1 - t) : t;   // un-reverse backward dir
            hs[(size_t)tout * HD + row] = h;
            hb[((t + 1) & 1) * 512 + row] = f2bf(h);
        }
        __threadfence();
        __syncthreads();
        // Cluster-wide split barrier: all 16 WGs exchange h before next step.
        __builtin_amdgcn_s_cluster_barrier();
    }
}

// ------------------------- classifier head -------------------------
__global__ void __launch_bounds__(256)
head_kernel(const float* __restrict__ h_fwd, const float* __restrict__ h_bwd,
            const float* __restrict__ W_out, const float* __restrict__ b_out,
            float* __restrict__ out) {
    int t = blockIdx.x * blockDim.x + threadIdx.x;
    if (t >= TLEN) return;
    const float* hf = h_fwd + (size_t)t * HD;
    const float* hb = h_bwd + (size_t)t * HD;
    float tag[NC];
#pragma unroll
    for (int c = 0; c < NC; ++c) {
        const float* wr = W_out + (size_t)c * (2 * HD);
        float s = b_out[c];
        for (int k = 0; k < HD; k += 4) {
            float4 hv = *(const float4*)(hf + k);
            float4 wv = *(const float4*)(wr + k);
            s += hv.x * wv.x + hv.y * wv.y + hv.z * wv.z + hv.w * wv.w;
            float4 hv2 = *(const float4*)(hb + k);
            float4 wv2 = *(const float4*)(wr + HD + k);
            s += hv2.x * wv2.x + hv2.y * wv2.y + hv2.z * wv2.z + hv2.w * wv2.w;
        }
        tag[c] = s;
    }
    float m   = fmaxf(tag[0], tag[1]);
    float lse = m + __logf(__expf(tag[0] - m) + __expf(tag[1] - m));
    out[(size_t)t * NC + 0] = tag[0] - lse;
    out[(size_t)t * NC + 1] = tag[1] - lse;
}

// ------------------------- launcher -------------------------
extern "C" void kernel_launch(void* const* d_in, const int* in_sizes, int n_in,
                              void* d_out, int out_size, void* d_ws, size_t ws_size,
                              hipStream_t stream) {
    (void)in_sizes; (void)n_in; (void)out_size; (void)ws_size;

    const int*   text  = (const int*)d_in[0];
    const float* emb   = (const float*)d_in[1];
    const float* Wih_f = (const float*)d_in[2];
    const float* Whh_f = (const float*)d_in[3];
    const float* bih_f = (const float*)d_in[4];
    const float* bhh_f = (const float*)d_in[5];
    const float* Wih_b = (const float*)d_in[6];
    const float* Whh_b = (const float*)d_in[7];
    const float* bih_b = (const float*)d_in[8];
    const float* bhh_b = (const float*)d_in[9];
    const float* W_out = (const float*)d_in[10];
    const float* b_out = (const float*)d_in[11];
    float* out = (float*)d_out;

    char* ws = (char*)d_ws;
    size_t off = 0;
    auto alloc = [&](size_t bytes) -> void* {
        void* p = ws + off;
        off = (off + bytes + 255) & ~(size_t)255;
        return p;
    };

    __bf16* X_bf     = (__bf16*)alloc((size_t)TLEN * ED * 2);
    __bf16* Wih_f_bf = (__bf16*)alloc((size_t)H4 * ED * 2);
    __bf16* Whh_f_bf = (__bf16*)alloc((size_t)H4 * HD * 2);
    __bf16* Wih_b_bf = (__bf16*)alloc((size_t)H4 * ED * 2);
    __bf16* Whh_b_bf = (__bf16*)alloc((size_t)H4 * HD * 2);
    float*  bsum_f   = (float*)alloc((size_t)H4 * 4);
    float*  bsum_b   = (float*)alloc((size_t)H4 * 4);
    float*  XP_f     = (float*)alloc((size_t)TLEN * H4 * 4);
    float*  XP_b     = (float*)alloc((size_t)TLEN * H4 * 4);
    float*  h_fwd    = (float*)alloc((size_t)TLEN * HD * 4);
    float*  h_bwd    = (float*)alloc((size_t)TLEN * HD * 4);
    __bf16* hbuf     = (__bf16*)alloc((size_t)2 * 2 * 512 * 2);

    const int WN = H4 * ED;   // 1048576 elements per weight matrix

    // 1) weight conversion f32 -> bf16 (L2-resident working set)
    f32_to_bf16_kernel<<<2048, 256, 0, stream>>>(Wih_f, Wih_f_bf, WN);
    f32_to_bf16_kernel<<<2048, 256, 0, stream>>>(Whh_f, Whh_f_bf, WN);
    f32_to_bf16_kernel<<<2048, 256, 0, stream>>>(Wih_b, Wih_b_bf, WN);
    f32_to_bf16_kernel<<<2048, 256, 0, stream>>>(Whh_b, Whh_b_bf, WN);
    bias_sum_kernel<<<8, 256, 0, stream>>>(bih_f, bhh_f, bsum_f, H4);
    bias_sum_kernel<<<8, 256, 0, stream>>>(bih_b, bhh_b, bsum_b, H4);
    zero_bf16_kernel<<<8, 256, 0, stream>>>(hbuf, 2 * 2 * 512);

    // 2) embedding gather -> bf16 activations
    gather_embed_kernel<<<TLEN, 128, 0, stream>>>(text, emb, X_bf);

    // 3) input-projection GEMMs (WMMA bf16 + async LDS staging); bwd reversed
    dim3 ggrid(H4 / 64, TLEN / 128);
    xproj_gemm_kernel<<<ggrid, 256, 0, stream>>>(X_bf, Wih_f_bf, bsum_f, XP_f, 0);
    xproj_gemm_kernel<<<ggrid, 256, 0, stream>>>(X_bf, Wih_b_bf, bsum_b, XP_b, 1);

    // 4) bidirectional recurrence: 16-WG cluster, cluster barrier per step
    bilstm_scan_kernel<<<16, 256, 0, stream>>>(Whh_f_bf, Whh_b_bf, XP_f, XP_b,
                                               h_fwd, h_bwd, hbuf);

    // 5) classifier head + log_softmax
    head_kernel<<<TLEN / 256, 256, 0, stream>>>(h_fwd, h_bwd, W_out, b_out, out);
}
